// MoELayer_10307921510926
// MI455X (gfx1250) — matrix operand order repaired and checked by reading, not compiled, not run on hardware
//
#include <hip/hip_runtime.h>
#include <hip/hip_bf16.h>
#include <stdint.h>

// Problem constants (B=8, S=2048, D=1024, E=8, F=4096)
#define NTOK   16384
#define DDIM   1024
#define EEXP   8
#define FDIM   4096

#define MT     32          // tokens per block tile (2 WMMA M-tiles)
#define FC     512         // F chunk of h staged in LDS per iteration
#define KB     256         // K block of x staged in LDS at a time
#define XSTR   (KB + 8)    // padded bf16 row stride for x block in LDS
#define HSTR   (FC + 8)    // padded bf16 row stride for h chunk in LDS

#define NT1    (FDIM / 16) // 256 B n-tiles in packed W1
#define NT2    (DDIM / 16) // 64  B n-tiles in packed W2

typedef __attribute__((ext_vector_type(16))) __bf16 bf16x16;
typedef __attribute__((ext_vector_type(8)))  float  floatx8;

// fp32 -> bf16 round-to-nearest-even
static __device__ __forceinline__ __bf16 f2bf(float f) {
  uint32_t u = __builtin_bit_cast(uint32_t, f);
  u += 0x7FFFu + ((u >> 16) & 1u);
  uint16_t h = (uint16_t)(u >> 16);
  return __builtin_bit_cast(__bf16, h);
}

struct V16B { uint4 q0, q1; };  // 32B = one lane's bf16 WMMA fragment

// A-fragment (16x32 bf16) from LDS: lane row M = lane&15.
// element i -> K = k0 + 16*g + i
static __device__ __forceinline__ bf16x16 ld_a_lds(const __bf16* row, int k0, int g) {
  V16B v;
  v.q0 = *(const uint4*)(row + k0 + 16 * g);
  v.q1 = *(const uint4*)(row + k0 + 16 * g + 8);
  return __builtin_bit_cast(bf16x16, v);
}

// B-fragment from pre-packed bf16 weights: one contiguous 32B record per lane.
static __device__ __forceinline__ bf16x16 ld_b_packed(const __bf16* wb, int kt, int NT,
                                                      int ntg, int lane) {
  const V16B* p = (const V16B*)wb;
  V16B v = p[(size_t)(kt * NT + ntg) * 32 + lane];
  return __builtin_bit_cast(bf16x16, v);
}

// Fallback B-fragment from row-major fp32 [K x ld]: lane holds column `col`;
// element i -> K = k0 + 16*g + i.
static __device__ __forceinline__ bf16x16 ld_b_f32(const float* base, int k0, int ld,
                                                   int col, int g) {
  bf16x16 b;
  const float* p = base + (size_t)(k0 + 16 * g) * ld + col;
#pragma unroll
  for (int i = 0; i < 16; ++i) b[i] = f2bf(p[(size_t)i * ld]);
  return b;
}

static __device__ __forceinline__ floatx8 wmma_bf16(bf16x16 a, bf16x16 b, floatx8 c) {
  return __builtin_amdgcn_wmma_f32_16x16x32_bf16(false, a, false, b, (short)0, c,
                                                 false, false);
}

// ---------------------------------------------------------------------------
// Kernel 0: pack W1/W2 fp32 -> bf16 in WMMA B-fragment order. One wave per
// 32x16 tile; lane writes its 32B fragment record coalesced.
// ---------------------------------------------------------------------------
__global__ void __launch_bounds__(512) pack_weights(const float* __restrict__ W1,
                                                    const float* __restrict__ W2,
                                                    __bf16* __restrict__ w1b,
                                                    __bf16* __restrict__ w2b) {
  const int lane = threadIdx.x & 31;
  const int wv   = blockIdx.x * 16 + (threadIdx.x >> 5);
  const int g = lane >> 4, n = lane & 15;

  if (wv < EEXP * (DDIM / 32) * NT1) {           // W1 tiles: 65536
    int e  = wv >> 13;                           // / 8192 tiles per expert
    int r  = wv & 8191;
    int kt = r >> 8;                             // / NT1
    int nt = r & (NT1 - 1);
    const float* src =
        W1 + (size_t)e * DDIM * FDIM + (size_t)(kt * 32 + 16 * g) * FDIM + nt * 16 + n;
    __bf16* dst =
        w1b + (size_t)e * DDIM * FDIM + ((size_t)(kt * NT1 + nt) * 32 + lane) * 16;
    union { __bf16 h[16]; V16B v; } u;
#pragma unroll
    for (int i = 0; i < 16; ++i) u.h[i] = f2bf(src[(size_t)i * FDIM]);
    *(V16B*)dst = u.v;
  } else {                                       // W2 tiles: 65536
    int t  = wv - EEXP * (DDIM / 32) * NT1;
    int e  = t >> 13;
    int r  = t & 8191;
    int kt = r >> 6;                             // / NT2 (F/32=128 kt)
    int nt = r & (NT2 - 1);
    const float* src =
        W2 + (size_t)e * FDIM * DDIM + (size_t)(kt * 32 + 16 * g) * DDIM + nt * 16 + n;
    __bf16* dst =
        w2b + (size_t)e * FDIM * DDIM + ((size_t)(kt * NT2 + nt) * 32 + lane) * 16;
    union { __bf16 h[16]; V16B v; } u;
#pragma unroll
    for (int i = 0; i < 16; ++i) u.h[i] = f2bf(src[(size_t)i * DDIM]);
    *(V16B*)dst = u.v;
  }
}

// ---------------------------------------------------------------------------
// Kernel 1: top-2 gating, one wave32 per token; appends to per-expert lists.
// ---------------------------------------------------------------------------
__global__ void __launch_bounds__(256) moe_gate(const float* __restrict__ x,
                                                const float* __restrict__ gw,
                                                const float* __restrict__ gb,
                                                int* __restrict__ cnt,
                                                int* __restrict__ tok_arr,
                                                float* __restrict__ w_arr) {
  const int lane = threadIdx.x & 31;
  const int wave = threadIdx.x >> 5;
  const int t = blockIdx.x * 8 + wave;
  if (t >= NTOK) return;

  float acc[EEXP];
#pragma unroll
  for (int e = 0; e < EEXP; ++e) acc[e] = 0.f;

  const float* xr = x + (size_t)t * DDIM;
  for (int i = 0; i < DDIM / 32; ++i) {
    int d = lane + 32 * i;
    float xv = xr[d];
    const float4* g4 = (const float4*)(gw + (size_t)d * EEXP);
    float4 a = g4[0], b = g4[1];
    acc[0] += xv * a.x; acc[1] += xv * a.y; acc[2] += xv * a.z; acc[3] += xv * a.w;
    acc[4] += xv * b.x; acc[5] += xv * b.y; acc[6] += xv * b.z; acc[7] += xv * b.w;
  }
#pragma unroll
  for (int e = 0; e < EEXP; ++e) {
    float v = acc[e];
    for (int s = 16; s > 0; s >>= 1) v += __shfl_xor(v, s, 32);
    acc[e] = v + gb[e];
  }
  if (lane == 0) {
    int i0 = 0;
#pragma unroll
    for (int e = 1; e < EEXP; ++e) if (acc[e] > acc[i0]) i0 = e;
    int i1 = (i0 == 0) ? 1 : 0;
#pragma unroll
    for (int e = 0; e < EEXP; ++e) if (e != i0 && acc[e] > acc[i1]) i1 = e;
    float w0 = 1.f / (1.f + __expf(acc[i1] - acc[i0]));
    float w1 = 1.f - w0;
    int p0 = atomicAdd(&cnt[i0], 1);
    tok_arr[i0 * NTOK + p0] = t;  w_arr[i0 * NTOK + p0] = w0;
    int p1 = atomicAdd(&cnt[i1], 1);
    tok_arr[i1 * NTOK + p1] = t;  w_arr[i1 * NTOK + p1] = w1;
  }
}

// ---------------------------------------------------------------------------
// Kernel 2: fused expert FFN, 32-token tile, 512 threads (16 waves).
// Per k-step: A loaded once from LDS, all tile B-fragments issued as a load
// clause, then 4/8 independent back-to-back WMMAs (shared A, distinct accs).
// ---------------------------------------------------------------------------
template <bool PACKED>
__global__ void __launch_bounds__(512) moe_ffn(const float* __restrict__ x,
                                               const float* __restrict__ W1,
                                               const float* __restrict__ b1,
                                               const float* __restrict__ W2,
                                               const float* __restrict__ b2,
                                               const __bf16* __restrict__ w1b,
                                               const __bf16* __restrict__ w2b,
                                               const int* __restrict__ cnt,
                                               const int* __restrict__ tok_arr,
                                               const float* __restrict__ w_arr,
                                               float* __restrict__ out) {
  __shared__ __align__(16) __bf16 x_lds[MT * XSTR];  // ~16.9 KB
  __shared__ __align__(16) __bf16 h_lds[MT * HSTR];  // ~33.3 KB
  __shared__ int   s_tok[MT];
  __shared__ float s_w[MT];

  const int e    = blockIdx.x / (NTOK / MT);
  const int tile = blockIdx.x % (NTOK / MT);
  const int n_e  = cnt[e];
  if (tile * MT >= n_e) return;   // block-uniform early exit

  const int tid  = threadIdx.x;
  const int lane = tid & 31;
  const int wave = tid >> 5;
  const int g    = lane >> 4;
  const int n    = lane & 15;

  // A wave's tiles share one M-tile in both GEMMs:
  const int mt   = wave >> 3;          // GEMM1: (wave*4)>>5 == GEMM2: (wave*8)>>6
  const int nb1  = (wave * 4) & 31;    // GEMM1 first n-tile (of 4)
  const int nb2  = (wave * 8) & 63;    // GEMM2 first n-tile (of 8)

  if (tid < MT) {
    int idx = tile * MT + tid;
    if (idx < n_e) { s_tok[tid] = tok_arr[e * NTOK + idx]; s_w[tid] = w_arr[e * NTOK + idx]; }
    else           { s_tok[tid] = -1;                      s_w[tid] = 0.f; }
  }

  const float*  W1e  = W1  + (size_t)e * DDIM * FDIM;
  const float*  W2e  = W2  + (size_t)e * FDIM * DDIM;
  const __bf16* w1be = PACKED ? w1b + (size_t)e * DDIM * FDIM : nullptr;
  const __bf16* w2be = PACKED ? w2b + (size_t)e * FDIM * DDIM : nullptr;
  const float*  b1e  = b1 + (size_t)e * FDIM;
  const float*  b2e  = b2 + (size_t)e * DDIM;

  const __bf16* arow = x_lds + (mt * 16 + n) * XSTR;
  const __bf16* hrow = h_lds + (mt * 16 + n) * HSTR;

  floatx8 accY[8];
#pragma unroll
  for (int j = 0; j < 8; ++j) accY[j] = (floatx8){0.f, 0.f, 0.f, 0.f, 0.f, 0.f, 0.f, 0.f};

  for (int ch = 0; ch < FDIM / FC; ++ch) {
    const int fbase = ch * FC;

    // ---- GEMM1: h_chunk[MT][FC] = relu(x[MT][D] @ W1[:, chunk] + b1) ----
    floatx8 acc1[4];
#pragma unroll
    for (int j = 0; j < 4; ++j) acc1[j] = (floatx8){0.f, 0.f, 0.f, 0.f, 0.f, 0.f, 0.f, 0.f};

    for (int kb = 0; kb < DDIM / KB; ++kb) {
      __syncthreads();  // prev compute done (incl. prev-chunk GEMM2 h reads)
      // stage x[MT][KB] fp32 -> bf16
      for (int i = tid; i < MT * KB; i += 512) {
        int r = i >> 8;             // / KB
        int c = i & (KB - 1);
        int tok = s_tok[r];
        float v = (tok >= 0) ? x[(size_t)tok * DDIM + kb * KB + c] : 0.f;
        x_lds[r * XSTR + c] = f2bf(v);
      }
      __syncthreads();

#pragma unroll 1
      for (int k0 = 0; k0 < KB; k0 += 32) {
        const int kt = (kb * KB + k0) >> 5;
        bf16x16 a = ld_a_lds(arow, k0, g);   // one A per k-step, shared by 4 tiles
        bf16x16 b[4];
#pragma unroll
        for (int j = 0; j < 4; ++j) {
          if (PACKED) b[j] = ld_b_packed(w1be, kt, NT1, (fbase >> 4) + nb1 + j, lane);
          else        b[j] = ld_b_f32(W1e, kb * KB + k0, FDIM, fbase + (nb1 + j) * 16 + n, g);
        }
#pragma unroll
        for (int j = 0; j < 4; ++j) acc1[j] = wmma_bf16(a, b[j], acc1[j]);
      }
    }

    // write h chunk (relu + bias) to LDS
#pragma unroll
    for (int j = 0; j < 4; ++j) {
      const int nt = nb1 + j;
      const float bias = b1e[fbase + nt * 16 + n];
#pragma unroll
      for (int r = 0; r < 8; ++r) {
        int m = mt * 16 + r + 8 * g;
        float v = acc1[j][r] + bias;
        h_lds[m * HSTR + nt * 16 + n] = f2bf(v > 0.f ? v : 0.f);
      }
    }
    __syncthreads();  // h chunk visible to all waves

    // ---- GEMM2: y[MT][D] += h_chunk @ W2[chunk, :] ----
#pragma unroll 1
    for (int k0 = 0; k0 < FC; k0 += 32) {
      const int kt = (fbase + k0) >> 5;
      bf16x16 a = ld_a_lds(hrow, k0, g);     // one A per k-step, shared by 8 tiles
      bf16x16 b[8];
#pragma unroll
      for (int j = 0; j < 8; ++j) {
        if (PACKED) b[j] = ld_b_packed(w2be, kt, NT2, nb2 + j, lane);
        else        b[j] = ld_b_f32(W2e + (size_t)fbase * DDIM, k0, DDIM, (nb2 + j) * 16 + n, g);
      }
#pragma unroll
      for (int j = 0; j < 8; ++j) accY[j] = wmma_bf16(a, b[j], accY[j]);
    }
  }

  // ---- Epilogue: + b2, gate-weight scale, atomic scatter ----
#pragma unroll
  for (int j = 0; j < 8; ++j) {
    const int dcol = (nb2 + j) * 16 + n;
    const float bias = b2e[dcol];
#pragma unroll
    for (int r = 0; r < 8; ++r) {
      int m = mt * 16 + r + 8 * g;
      int tok = s_tok[m];
      if (tok >= 0)
        atomicAdd(out + (size_t)tok * DDIM + dcol, s_w[m] * (accY[j][r] + bias));
    }
  }
}

// ---------------------------------------------------------------------------
extern "C" void kernel_launch(void* const* d_in, const int* in_sizes, int n_in,
                              void* d_out, int out_size, void* d_ws, size_t ws_size,
                              hipStream_t stream) {
  (void)in_sizes; (void)n_in;
  const float* x  = (const float*)d_in[0];
  const float* gw = (const float*)d_in[1];
  const float* gb = (const float*)d_in[2];
  const float* W1 = (const float*)d_in[3];
  const float* b1 = (const float*)d_in[4];
  const float* W2 = (const float*)d_in[5];
  const float* b2 = (const float*)d_in[6];
  float* out = (float*)d_out;

  // ws layout: [0,64) counters | tok int[E*N] | w float[E*N] | (opt) packed bf16 W1,W2
  int*   cnt     = (int*)d_ws;
  int*   tok_arr = (int*)((char*)d_ws + 64);
  float* w_arr   = (float*)((char*)d_ws + 64 + sizeof(int) * (size_t)EEXP * NTOK);

  size_t route_bytes  = 64 + 8ull * EEXP * NTOK;
  size_t packed_off   = (route_bytes + 255) & ~(size_t)255;
  size_t packed_bytes = 2ull * sizeof(__bf16) * EEXP * DDIM * FDIM;  // W1b + W2b
  bool   packed       = ws_size >= packed_off + packed_bytes;
  __bf16* w1b = (__bf16*)((char*)d_ws + packed_off);
  __bf16* w2b = w1b + (size_t)EEXP * DDIM * FDIM;

  hipMemsetAsync(cnt, 0, 64, stream);
  hipMemsetAsync(out, 0, (size_t)out_size * sizeof(float), stream);

  if (packed)
    pack_weights<<<8192, 512, 0, stream>>>(W1, W2, w1b, w2b);

  moe_gate<<<NTOK / 8, 256, 0, stream>>>(x, gw, gb, cnt, tok_arr, w_arr);

  if (packed)
    moe_ffn<true><<<EEXP * (NTOK / MT), 512, 0, stream>>>(
        x, W1, b1, W2, b2, w1b, w2b, cnt, tok_arr, w_arr, out);
  else
    moe_ffn<false><<<EEXP * (NTOK / MT), 512, 0, stream>>>(
        x, W1, b1, W2, b2, nullptr, nullptr, cnt, tok_arr, w_arr, out);
}
